// GeneralizedLiquidNeuralNetwork_38001870635353
// MI455X (gfx1250) — matrix operand order repaired and checked by reading, not compiled, not run on hardware
//
#include <hip/hip_runtime.h>
#include <hip/hip_bf16.h>
#include <math.h>

// ---------------------------------------------------------------------------
// GeneralizedLiquidNeuralNetwork for MI455X (gfx1250).
// Sequential LTC scan -> single persistent workgroup (32 waves on one WGP),
// state in LDS (320KB/WGP), params streamed via async global->LDS (ASYNCcnt),
// final 8x64 @ 64x8 projection done with v_wmma_f32_16x16x32_f16 using
// zero-padded f16 LDS tiles so the packing is branch-free.
// ---------------------------------------------------------------------------

#define EPSV 1e-8f

typedef __attribute__((ext_vector_type(16))) _Float16 v16h;
typedef __attribute__((ext_vector_type(8)))  float    v8f;

__device__ __forceinline__ float softplusf(float x) {
  // log(1 + exp(x)), numerically stable
  return fmaxf(x, 0.0f) + log1pf(__expf(-fabsf(x)));
}
__device__ __forceinline__ float sigmoidf_(float x) {
  return 1.0f / (1.0f + __expf(-x));
}

// ---------------------------------------------------------------------------
// Kernel 1: fuse parameters (parallel, runs once).
//   CJ[j,k]  = softplus(w)*mask*erev      (num coeff; den coeff = |CJ|)
//   SC[s,k]  = softplus(sw)*smask*serev
//   glv[k]   = softplus(gleak); glvl[k] = glv*vleak; cmv[k] = softplus(cm)
// ---------------------------------------------------------------------------
__global__ void glnn_precompute(const float* __restrict__ w,
                                const float* __restrict__ mask,
                                const float* __restrict__ erev,
                                const float* __restrict__ sw,
                                const float* __restrict__ smask,
                                const float* __restrict__ serev,
                                const float* __restrict__ gleak,
                                const float* __restrict__ vleak,
                                const float* __restrict__ cm,
                                float* __restrict__ CJ,
                                float* __restrict__ SC,
                                float* __restrict__ glv,
                                float* __restrict__ glvl,
                                float* __restrict__ cmv) {
  int t = blockIdx.x * blockDim.x + threadIdx.x;
  if (t < 256 * 256) CJ[t] = softplusf(w[t]) * mask[t] * erev[t];
  if (t < 32 * 256)  SC[t] = softplusf(sw[t]) * smask[t] * serev[t];
  if (t < 256) {
    float g = softplusf(gleak[t]);
    glv[t]  = g;
    glvl[t] = g * vleak[t];
    cmv[t]  = softplusf(cm[t]);
  }
}

// ---------------------------------------------------------------------------
// Kernel 2: persistent single-workgroup LTC scan.
// 1024 threads: thread = (k = tid&255, lg = tid>>8); each owns 4 (l,k) cells,
// l = lg*4 + i.  Dynamic LDS layout (bytes):
// ---------------------------------------------------------------------------
#define OFF_VE   0u        // eval state v[16][256]              16 KB
#define OFF_WNS  16384u    // sensory num sums [16][256]         16 KB
#define OFF_WDS  32768u    // sensory den sums [16][256]         16 KB
#define OFF_XIN  49152u    // xin [16][32] f32                    2 KB
#define OFF_OUTS 51200u    // outs_last f16 [16][64] (pad rows)   2 KB
#define OFF_MISC 53248u    // tsort[16], pos, raw ts[16]        256 B
#define OFF_WPAD 53504u    // Wout staged f16 [64][16] (pad cols) 2 KB
#define OFF_TC   57344u    // CJ tile [32][256]                  32 KB
#define OFF_TM   90112u    // mu tile                            32 KB
#define OFF_TS2  122880u   // sigma tile                         32 KB
#define SMEM_BYTES 155648u

__global__ __launch_bounds__(1024)
void glnn_scan(const float* __restrict__ x,
               const float* __restrict__ input_w, const float* __restrict__ input_b,
               const float* __restrict__ smu, const float* __restrict__ ssig,
               const float* __restrict__ mu, const float* __restrict__ sigma,
               const float* __restrict__ output_w, const float* __restrict__ output_b,
               const float* __restrict__ Wout, const float* __restrict__ bout,
               const float* __restrict__ CJ, const float* __restrict__ SC,
               const float* __restrict__ glv, const float* __restrict__ glvl,
               const float* __restrict__ cmv,
               float* __restrict__ out) {
  extern __shared__ float smem[];
  float*     vE     = smem + OFF_VE / 4;
  float*     wns    = smem + OFF_WNS / 4;
  float*     wds    = smem + OFF_WDS / 4;
  float*     xin    = smem + OFF_XIN / 4;
  _Float16*  outsLh = (_Float16*)(smem + OFF_OUTS / 4);   // [16][64], rows 8..15 zero
  float*     misc   = smem + OFF_MISC / 4;                // [0..15] tsort, raw ts [32..47]
  int*       miscI  = (int*)(misc + 16);                  // [0] = pos
  _Float16*  wouth  = (_Float16*)(smem + OFF_WPAD / 4);   // [64][16], cols 8..15 zero
  float* tC = smem + OFF_TC / 4;
  float* tM = smem + OFF_TM / 4;
  float* tS = smem + OFF_TS2 / 4;

  const int tid = threadIdx.x;
  const int k  = tid & 255;
  const int lg = tid >> 8;
  const float glvk = glv[k], glvlk = glvl[k], cmvk = cmv[k];

  // One-time init: zero-padded WMMA operand tiles (f16 in LDS).
  // outsLh: 16*64 = 1024 halves; wouth: 64*16 = 1024 halves. One per thread.
  outsLh[tid] = (_Float16)0.0f;
  {
    int m = tid >> 4, o = tid & 15;
    wouth[tid] = (o < 8) ? (_Float16)Wout[m * 8 + o] : (_Float16)0.0f;
  }

  float hv[4] = {0.f, 0.f, 0.f, 0.f};  // carried state h for (l=lg*4+i, k)

  // dyn(): requires eval state already stored in vE by all threads.
  auto dynEval = [&](float tcur, const float veval[4], float dres[4]) {
    float ns[4] = {0.f, 0.f, 0.f, 0.f}, ds[4] = {0.f, 0.f, 0.f, 0.f};
    const unsigned goffB = (unsigned)tid * 32u;  // 8 floats per thread per tile
    for (int jt = 0; jt < 8; ++jt) {
      __syncthreads();  // prior tile fully consumed; vE writes published
      const float* gC = CJ    + jt * 8192;
      const float* gM = mu    + jt * 8192;
      const float* gS = sigma + jt * 8192;
      unsigned dC = OFF_TC  + goffB;
      unsigned dM = OFF_TM  + goffB;
      unsigned dS = OFF_TS2 + goffB;
      // async global -> LDS, 2x b128 per matrix per thread (ASYNCcnt path)
      asm volatile("global_load_async_to_lds_b128 %0, %1, %2\n\t"
                   "global_load_async_to_lds_b128 %0, %1, %2 offset:16"
                   :: "v"(dC), "v"(goffB), "s"(gC) : "memory");
      asm volatile("global_load_async_to_lds_b128 %0, %1, %2\n\t"
                   "global_load_async_to_lds_b128 %0, %1, %2 offset:16"
                   :: "v"(dM), "v"(goffB), "s"(gM) : "memory");
      asm volatile("global_load_async_to_lds_b128 %0, %1, %2\n\t"
                   "global_load_async_to_lds_b128 %0, %1, %2 offset:16"
                   :: "v"(dS), "v"(goffB), "s"(gS) : "memory");
      asm volatile("s_wait_asynccnt 0x0" ::: "memory");
      __syncthreads();  // whole tile resident
      for (int jj = 0; jj < 32; ++jj) {
        float c = tC[jj * 256 + k];
        float m = tM[jj * 256 + k];
        float g = tS[jj * 256 + k];
        float ac = fabsf(c);
        int jgl = jt * 32 + jj;
#pragma unroll
        for (int i = 0; i < 4; ++i) {
          float vv = vE[(lg * 4 + i) * 256 + jgl];  // wave-broadcast read
          float sg = sigmoidf_((vv - m) * g);
          ns[i] += c * sg;
          ds[i] += ac * sg;
        }
      }
    }
    __syncthreads();  // all readers of vE done before caller overwrites it
    float cmt = cmvk / (tcur + EPSV);
#pragma unroll
    for (int i = 0; i < 4; ++i) {
      int lk = (lg * 4 + i) * 256 + k;
      float num = cmt * veval[i] + glvlk + ns[i] + wns[lk];
      float den = cmt + glvk + ds[i] + wds[lk];
      dres[i] = num / (den + EPSV) - veval[i];
    }
  };

  auto writeEval = [&](const float ve[4]) {
#pragma unroll
    for (int i = 0; i < 4; ++i) vE[(lg * 4 + i) * 256 + k] = ve[i];
  };

  for (int b = 0; b < 8; ++b) {
    // stage raw ts and compute xin = inp*input_w + input_b
    if (tid < 16) misc[32 + tid] = x[b * 528 + tid * 33];
    if (tid < 512) {
      int l = tid >> 5, s = tid & 31;
      xin[tid] = x[b * 528 + l * 33 + 1 + s] * input_w[s] + input_b[s];
    }
    __syncthreads();
    if (tid == 0) {
      // stable argsort of 16 times (rank, index tie-break); pos = rank(ts[15])
      float tsv[16];
      for (int l = 0; l < 16; ++l) tsv[l] = misc[32 + l];
      int posv = 0;
      for (int l = 0; l < 16; ++l) {
        int r = 0;
        for (int l2 = 0; l2 < 16; ++l2)
          if (tsv[l2] < tsv[l] || (tsv[l2] == tsv[l] && l2 < l)) ++r;
        misc[r] = tsv[l];
        if (l == 15) posv = r;
      }
      miscI[0] = posv;
    }
    // sensory synapse sums wns/wds (each thread writes only its own cells)
    float wnsA[4] = {0.f, 0.f, 0.f, 0.f}, wdsA[4] = {0.f, 0.f, 0.f, 0.f};
    for (int s = 0; s < 32; ++s) {
      float c = SC[s * 256 + k], m = smu[s * 256 + k], g = ssig[s * 256 + k];
      float ac = fabsf(c);
#pragma unroll
      for (int i = 0; i < 4; ++i) {
        float sg = sigmoidf_((xin[(lg * 4 + i) * 32 + s] - m) * g);
        wnsA[i] += c * sg;
        wdsA[i] += ac * sg;
      }
    }
#pragma unroll
    for (int i = 0; i < 4; ++i) {
      wns[(lg * 4 + i) * 256 + k] = wnsA[i];
      wds[(lg * 4 + i) * 256 + k] = wdsA[i];
    }
    __syncthreads();
    int pos = miscI[0];  // uniform across block

    // new_state = v after exactly `pos` RK steps (v_out[pos]); steps beyond
    // pos never influence the result, so run only pos of them.
    for (int st = 0; st < pos; ++st) {
      float t0 = misc[st], t1 = misc[st + 1], dt = t1 - t0;
      float ve[4], k1[4], k2[4], k3[4], k4[4];
#pragma unroll
      for (int i = 0; i < 4; ++i) ve[i] = hv[i];
      writeEval(ve);  dynEval(t0, ve, k1);
#pragma unroll
      for (int i = 0; i < 4; ++i) ve[i] = hv[i] + dt * k1[i] * (1.0f / 3.0f);
      writeEval(ve);  dynEval(t0 + dt * (1.0f / 3.0f), ve, k2);
#pragma unroll
      for (int i = 0; i < 4; ++i) ve[i] = hv[i] + dt * (k2[i] - k1[i] * (1.0f / 3.0f));
      writeEval(ve);  dynEval(t0 + dt * (2.0f / 3.0f), ve, k3);
#pragma unroll
      for (int i = 0; i < 4; ++i) ve[i] = hv[i] + dt * (k1[i] - k2[i] + k3[i]);
      writeEval(ve);  dynEval(t1, ve, k4);
#pragma unroll
      for (int i = 0; i < 4; ++i)
        hv[i] += dt * (k1[i] + 3.0f * k2[i] + 3.0f * k3[i] + k4[i]) * 0.125f;
    }

    // out row used downstream: new_state[l=15, 0:64]*output_w + output_b
    if (lg == 3 && k < 64)
      outsLh[b * 64 + k] = (_Float16)(hv[3] * output_w[k] + output_b[k]);
    __syncthreads();
  }

  // Final projection: (16x64 zero-padded) @ (64x16 zero-padded) via two
  // chained v_wmma_f32_16x16x32_f16 (K = 32 each). Wave 0 only; EXEC all
  // ones within the wave (wave32). All operand fetches are unconditional
  // LDS half-loads from the padded tiles -> branch-free packing.
  if (tid < 32) {
    const int lane = tid;
    const bool hi = lane >= 16;
    const int n16 = lane & 15;
    v8f acc = {0.f, 0.f, 0.f, 0.f, 0.f, 0.f, 0.f, 0.f};
#pragma unroll
    for (int kk = 0; kk < 2; ++kk) {
      v16h a, bm;
#pragma unroll
      for (int e = 0; e < 16; ++e) {
        // A (16x32, f16): lane holds row M=lane%16; ISA K-interleave layout
        int p = e >> 1, h = e & 1;
        int Ka = ((p & 3) * 2 + h) + ((p >= 4) ? 16 : 0) + (hi ? 8 : 0) + kk * 32;
        a[e] = outsLh[n16 * 64 + Ka];
        // B (32x16, f16): VGPR half e holds row K, lane holds col N=lane%16
        int Kb = (e & 7) + ((e >= 8) ? 16 : 0) + (hi ? 8 : 0) + kk * 32;
        bm[e] = wouth[Kb * 16 + n16];
      }
      acc = __builtin_amdgcn_wmma_f32_16x16x32_f16(
          false, a, false, bm, (short)0, acc, false, false);
    }
    // D layout: VGPR r = row M=r on lanes 0..15 (N = lane)
    if (!hi && n16 < 8) {
#pragma unroll
      for (int r = 0; r < 8; ++r) out[r * 8 + n16] = acc[r] + bout[n16];
    }
  }
}

// ---------------------------------------------------------------------------
extern "C" void kernel_launch(void* const* d_in, const int* in_sizes, int n_in,
                              void* d_out, int out_size, void* d_ws, size_t ws_size,
                              hipStream_t stream) {
  const float* x        = (const float*)d_in[0];
  const float* input_w  = (const float*)d_in[1];
  const float* input_b  = (const float*)d_in[2];
  const float* sw       = (const float*)d_in[3];
  const float* smu      = (const float*)d_in[4];
  const float* ssig     = (const float*)d_in[5];
  const float* serev    = (const float*)d_in[6];
  const float* smask    = (const float*)d_in[7];
  const float* w        = (const float*)d_in[8];
  const float* mu       = (const float*)d_in[9];
  const float* sigma    = (const float*)d_in[10];
  const float* erev     = (const float*)d_in[11];
  const float* mask     = (const float*)d_in[12];
  const float* gleak    = (const float*)d_in[13];
  const float* vleak    = (const float*)d_in[14];
  const float* cm       = (const float*)d_in[15];
  const float* output_w = (const float*)d_in[16];
  const float* output_b = (const float*)d_in[17];
  const float* Wout     = (const float*)d_in[18];
  const float* bout     = (const float*)d_in[19];

  float* ws   = (float*)d_ws;
  float* CJ   = ws;             // 65536
  float* SC   = ws + 65536;     // 8192
  float* glv  = ws + 73728;     // 256
  float* glvl = ws + 73984;     // 256
  float* cmv  = ws + 74240;     // 256

  glnn_precompute<<<256, 256, 0, stream>>>(w, mask, erev, sw, smask, serev,
                                           gleak, vleak, cm,
                                           CJ, SC, glv, glvl, cmv);

  (void)hipFuncSetAttribute((const void*)glnn_scan,
                            hipFuncAttributeMaxDynamicSharedMemorySize,
                            (int)SMEM_BYTES);
  glnn_scan<<<1, 1024, SMEM_BYTES, stream>>>(
      x, input_w, input_b, smu, ssig, mu, sigma, output_w, output_b,
      Wout, bout, CJ, SC, glv, glvl, cmv, (float*)d_out);
}
